// GCN_64338610094338
// MI455X (gfx1250) — matrix operand order repaired and checked by reading, not compiled, not run on hardware
//
#include <hip/hip_runtime.h>

// ---------------- WMMA fragment types (gfx1250, wave32) ----------------
typedef __attribute__((ext_vector_type(16))) __bf16 v16bf;
typedef __attribute__((ext_vector_type(8)))  float  v8f;

union FragBF { uint4 u4[2]; v16bf v; };

__device__ __forceinline__ unsigned f2bf_bits(float f) {
  unsigned u = __float_as_uint(f);
  u += 0x7FFFu + ((u >> 16) & 1u);     // round-to-nearest-even
  return u >> 16;
}

// ---------------- small utility kernels ----------------
__global__ void k_zero_f32(float* __restrict__ p, int n) {
  int i = blockIdx.x * blockDim.x + threadIdx.x;
  if (i < n) p[i] = 0.0f;
}

__global__ void k_deg_accum(const int* __restrict__ dst, const float* __restrict__ ew,
                            float* __restrict__ deg, int ne) {
  int i = blockIdx.x * blockDim.x + threadIdx.x;
  if (i < ne) atomicAdd(&deg[dst[i]], ew[i]);
}

__global__ void k_make_dis(float* __restrict__ deg_dis, int n) {
  int i = blockIdx.x * blockDim.x + threadIdx.x;
  if (i < n) deg_dis[i] = rsqrtf(deg_dis[i] + 1.0f);  // +1 = self loop; always > 0
}

// x_bf16[i][c] = bf16(emb[n_id[i]][c])
__global__ void k_gather_bf16(const float* __restrict__ emb, const int* __restrict__ n_id,
                              unsigned short* __restrict__ out, int n, int C) {
  int idx = blockIdx.x * blockDim.x + threadIdx.x;
  if (idx >= n * C) return;
  int i = idx / C, c = idx % C;
  out[idx] = (unsigned short)f2bf_bits(emb[(size_t)n_id[i] * C + c]);
}

// Pack W [K][N] row-major fp32 into WMMA B-fragment order (bf16 pairs),
// interleaved by groups of 4 N-tiles so one wave's per-K-step B block is
// a single contiguous 4KB chunk:
//   Wp[(((ng*KB + kb)*4 + j)*32 + lane)*8 + v]
//   K of pair = kb*32 + (v>>2)*16 + hi*8 + 2*(v&3),  N = (ng*4+j)*16 + (lane&15)
__global__ void k_packW(const float* __restrict__ W, unsigned* __restrict__ Wp, int K, int N) {
  int idx = blockIdx.x * blockDim.x + threadIdx.x;
  int KB = K >> 5;
  int total = (N >> 4) * KB * 32 * 8;   // == (N>>6)*KB*4*32*8
  if (idx >= total) return;
  int v    = idx & 7;
  int lane = (idx >> 3) & 31;
  int j    = (idx >> 8) & 3;
  int rem  = idx >> 10;
  int kb   = rem % KB;
  int ng   = rem / KB;
  int hi = lane >> 4;
  int n  = lane & 15;
  int k  = kb * 32 + (v >> 2) * 16 + hi * 8 + 2 * (v & 3);
  int col = (ng * 4 + j) * 16 + n;
  unsigned lo = f2bf_bits(W[(size_t)k * N + col]);
  unsigned hh = f2bf_bits(W[(size_t)(k + 1) * N + col]);
  Wp[idx] = lo | (hh << 16);
}

// ---------------- WMMA GEMM: Out[M,N] = Xbf[M,K] * W(packed)[K,N] ----------------
// One wave computes a 16x64 output strip (4 adjacent 16x16 tiles).
// Software-pipelined: fragments for K-step kb+1 are loaded before the WMMAs of kb.
// Requires N % 64 == 0 (true here: 256 and 128).
struct Frags { FragBF a, b0, b1, b2, b3; };

__device__ __forceinline__ void load_frags(Frags& f, const unsigned short* __restrict__ aptr,
                                           const unsigned char* __restrict__ bbase, int kb) {
  f.a.u4[0] = *(const uint4*)(aptr + kb * 32);        // K = kb*32 + hi*8 + 0..7
  f.a.u4[1] = *(const uint4*)(aptr + kb * 32 + 16);   // K = kb*32 + hi*8 + 16..23
  const unsigned char* bp = bbase + (size_t)kb * 4096;
  f.b0.u4[0] = *(const uint4*)(bp);
  f.b0.u4[1] = *(const uint4*)(bp + 16);
  f.b1.u4[0] = *(const uint4*)(bp + 1024);
  f.b1.u4[1] = *(const uint4*)(bp + 1040);
  f.b2.u4[0] = *(const uint4*)(bp + 2048);
  f.b2.u4[1] = *(const uint4*)(bp + 2064);
  f.b3.u4[0] = *(const uint4*)(bp + 3072);
  f.b3.u4[1] = *(const uint4*)(bp + 3088);
}

__global__ void k_gemm_bf16x4(const unsigned short* __restrict__ Xbf,
                              const unsigned* __restrict__ Wp,
                              float* __restrict__ Out, int M, int K, int N) {
  int lane = threadIdx.x & 31;
  int wid  = (blockIdx.x * blockDim.x + threadIdx.x) >> 5;
  int NG = N >> 6, KB = K >> 5, MT = M >> 4;
  int mt = wid / NG, ng = wid % NG;
  if (mt >= MT) return;
  int hi = lane >> 4, hl = lane & 15;

  // A (16x32 bf16): lane half 'hi' holds K = {hi*8..+7} and {16+hi*8..+7}
  const unsigned short* aptr = Xbf + (size_t)(mt * 16 + hl) * K + hi * 8;
  // B: one contiguous 4KB block per K-step for the whole wave
  const unsigned char* bbase = (const unsigned char*)Wp + (size_t)ng * KB * 4096
                             + (size_t)lane * 32;

  v8f c0 = {}, c1 = {}, c2 = {}, c3 = {};
  Frags fA, fB;

#define WMMA4(F)                                                                              \
  c0 = __builtin_amdgcn_wmma_f32_16x16x32_bf16(false, (F).a.v, false, (F).b0.v, (short)0, c0, \
                                               false, false);                                 \
  c1 = __builtin_amdgcn_wmma_f32_16x16x32_bf16(false, (F).a.v, false, (F).b1.v, (short)0, c1, \
                                               false, false);                                 \
  c2 = __builtin_amdgcn_wmma_f32_16x16x32_bf16(false, (F).a.v, false, (F).b2.v, (short)0, c2, \
                                               false, false);                                 \
  c3 = __builtin_amdgcn_wmma_f32_16x16x32_bf16(false, (F).a.v, false, (F).b3.v, (short)0, c3, \
                                               false, false)

  load_frags(fA, aptr, bbase, 0);
  int kb = 0;
  for (; kb + 2 <= KB; kb += 2) {
    load_frags(fB, aptr, bbase, kb + 1);
    WMMA4(fA);
    if (kb + 2 < KB) load_frags(fA, aptr, bbase, kb + 2);
    WMMA4(fB);
  }
  if (KB & 1) { WMMA4(fA); }
#undef WMMA4

  // C/D layout: VGPR r -> M = r + hi*8, N = hl (within each 16-wide tile)
  for (int r = 0; r < 8; ++r) {
    int row = mt * 16 + hi * 8 + r;
    float* orow = Out + (size_t)row * N + (size_t)ng * 64 + hl;
    orow[0]  = c0[r];
    orow[16] = c1[r];
    orow[32] = c2[r];
    orow[48] = c3[r];
  }
}

// agg[i][c] = dis[i]^2 * h[i][c]   (self-loop contribution; norm = dis[i]*1*dis[i])
__global__ void k_self_init(const float* __restrict__ h, const float* __restrict__ dis,
                            float* __restrict__ agg, int n, int C) {
  int idx = blockIdx.x * blockDim.x + threadIdx.x;
  if (idx >= n * C) return;
  int i = idx / C;
  float d = dis[i];
  agg[idx] = d * d * h[idx];
}

// one wave per edge: agg[dst] += dis[src]*ew*dis[dst] * h[src]
__global__ void k_scatter(const float* __restrict__ h, float* __restrict__ agg,
                          const int* __restrict__ src, const int* __restrict__ dst,
                          const float* __restrict__ ew, const float* __restrict__ dis,
                          int ne, int C) {
  int wid  = (blockIdx.x * blockDim.x + threadIdx.x) >> 5;
  int lane = threadIdx.x & 31;
  if (wid >= ne) return;
  int s = src[wid], d = dst[wid];
  float nrm = dis[s] * ew[wid] * dis[d];
  const float* hs = h + (size_t)s * C;
  float* ad = agg + (size_t)d * C;
  for (int c = lane; c < C; c += 32)
    atomicAdd(&ad[c], nrm * hs[c]);
}

// x1bf = bf16(relu(agg + b))
__global__ void k_bias_relu_bf16(const float* __restrict__ agg, const float* __restrict__ b,
                                 unsigned short* __restrict__ out, int n, int C) {
  int idx = blockIdx.x * blockDim.x + threadIdx.x;
  if (idx >= n * C) return;
  float v = agg[idx] + b[idx % C];
  v = v > 0.0f ? v : 0.0f;
  out[idx] = (unsigned short)f2bf_bits(v);
}

// z = agg + b (in place, fp32)
__global__ void k_bias(float* __restrict__ agg, const float* __restrict__ b, int n, int C) {
  int idx = blockIdx.x * blockDim.x + threadIdx.x;
  if (idx >= n * C) return;
  agg[idx] = agg[idx] + b[idx % C];
}

// one wave per label edge: out = dot(z[ls], z[ld]) over C channels
__global__ void k_dot(const float* __restrict__ z, const int* __restrict__ ls,
                      const int* __restrict__ ld, float* __restrict__ out, int nl, int C) {
  int wid  = (blockIdx.x * blockDim.x + threadIdx.x) >> 5;
  int lane = threadIdx.x & 31;
  if (wid >= nl) return;
  const float* a = z + (size_t)ls[wid] * C;
  const float* b = z + (size_t)ld[wid] * C;
  float s = 0.0f;
  for (int c = lane; c < C; c += 32) s += a[c] * b[c];
  for (int off = 16; off; off >>= 1) s += __shfl_xor(s, off, 32);
  if (lane == 0) out[wid] = s;
}

// ---------------- launch ----------------
static inline size_t alignup(size_t x) { return (x + 255) & ~(size_t)255; }

extern "C" void kernel_launch(void* const* d_in, const int* in_sizes, int n_in,
                              void* d_out, int out_size, void* d_ws, size_t ws_size,
                              hipStream_t stream) {
  const int*   n_id = (const int*)d_in[0];
  const int*   eidx = (const int*)d_in[1];       // [2][NE]
  const float* ew   = (const float*)d_in[2];
  const int*   elbl = (const int*)d_in[3];       // [2][NL]
  const float* emb  = (const float*)d_in[4];
  const float* W1   = (const float*)d_in[5];
  const float* b1   = (const float*)d_in[6];
  const float* W2   = (const float*)d_in[7];
  const float* b2   = (const float*)d_in[8];
  float* out = (float*)d_out;

  const int N   = in_sizes[0];              // 50000 nodes
  const int NE  = in_sizes[2];              // 800000 edges
  const int NL  = in_sizes[3] / 2;          // 100000 label pairs
  const int INC = in_sizes[4] / N;          // 256
  const int HID = in_sizes[6];              // 256
  const int OUT = in_sizes[8];              // 128

  const int* src = eidx;
  const int* dst = eidx + NE;
  const int* ls  = elbl;
  const int* ld  = elbl + NL;

  // ---- workspace carve-up ----
  char* w = (char*)d_ws;
  float* dis = (float*)w;            w += alignup((size_t)N * 4);
  unsigned short* xbf = (unsigned short*)w;  w += alignup((size_t)N * INC * 2);
  unsigned* w1p = (unsigned*)w;      w += alignup((size_t)(HID / 16) * (INC / 32) * 32 * 8 * 4);
  unsigned* w2p = (unsigned*)w;      w += alignup((size_t)(OUT / 16) * (HID / 32) * 32 * 8 * 4);
  float* hbuf = (float*)w;           w += alignup((size_t)N * HID * 4);   // h1, later h2
  float* aggbuf = (float*)w;         w += alignup((size_t)N * HID * 4);   // agg1, later agg2/z
  unsigned short* x1bf = (unsigned short*)w;  // N*HID bf16
  (void)ws_size; (void)n_in; (void)out_size;

  const int B = 256;

  // degree -> dis
  k_zero_f32<<<(N + B - 1) / B, B, 0, stream>>>(dis, N);
  k_deg_accum<<<(NE + B - 1) / B, B, 0, stream>>>(dst, ew, dis, NE);
  k_make_dis<<<(N + B - 1) / B, B, 0, stream>>>(dis, N);

  // bf16 input + packed weights
  k_gather_bf16<<<((size_t)N * INC + B - 1) / B, B, 0, stream>>>(emb, n_id, xbf, N, INC);
  {
    int t1 = (HID / 16) * (INC / 32) * 32 * 8;
    int t2 = (OUT / 16) * (HID / 32) * 32 * 8;
    k_packW<<<(t1 + B - 1) / B, B, 0, stream>>>(W1, w1p, INC, HID);
    k_packW<<<(t2 + B - 1) / B, B, 0, stream>>>(W2, w2p, HID, OUT);
  }

  // ---- layer 1 ----
  {
    int waves = (N / 16) * (HID / 64);
    k_gemm_bf16x4<<<(waves + 7) / 8, B, 0, stream>>>(xbf, w1p, hbuf, N, INC, HID);
  }
  k_self_init<<<((size_t)N * HID + B - 1) / B, B, 0, stream>>>(hbuf, dis, aggbuf, N, HID);
  k_scatter<<<(NE + 7) / 8, B, 0, stream>>>(hbuf, aggbuf, src, dst, ew, dis, NE, HID);
  k_bias_relu_bf16<<<((size_t)N * HID + B - 1) / B, B, 0, stream>>>(aggbuf, b1, x1bf, N, HID);

  // ---- layer 2 (hbuf/aggbuf reused) ----
  {
    int waves = (N / 16) * (OUT / 64);
    k_gemm_bf16x4<<<(waves + 7) / 8, B, 0, stream>>>(x1bf, w2p, hbuf, N, HID, OUT);
  }
  k_self_init<<<((size_t)N * OUT + B - 1) / B, B, 0, stream>>>(hbuf, dis, aggbuf, N, OUT);
  k_scatter<<<(NE + 7) / 8, B, 0, stream>>>(hbuf, aggbuf, src, dst, ew, dis, NE, OUT);
  k_bias<<<((size_t)N * OUT + B - 1) / B, B, 0, stream>>>(aggbuf, b2, N, OUT);

  // ---- link prediction dots ----
  k_dot<<<(NL + 7) / 8, B, 0, stream>>>(aggbuf, ls, ld, out, NL, OUT);
}
